// TopDown2LSTM_7748121002687
// MI455X (gfx1250) — matrix operand order repaired and checked by reading, not compiled, not run on hardware
//
#include <hip/hip_runtime.h>
#include <hip/hip_bf16.h>
#include <math.h>

// ---------------------------------------------------------------------------
// TopDown attention 2-layer LSTM greedy decoder for MI455X (gfx1250).
// All GEMMs: v_wmma_f32_16x16x32_bf16. Hot-loop GEMMs keep A/B resident in
// bf16 and stage tiles with GLOBAL_LOAD_ASYNC_TO_LDS_B128 (double-buffered,
// ASYNCcnt-tracked). Workspace requirement: ~149 MB (layout below).
// ---------------------------------------------------------------------------

typedef __attribute__((ext_vector_type(16))) __bf16 v16bf;
typedef __attribute__((ext_vector_type(8)))  float  v8f;
typedef __attribute__((ext_vector_type(4)))  int    v4i;

struct U4 { unsigned int a, b, c, d; };        // POD 16-byte chunk (union-safe)
union FragBF { v16bf v; U4 q[2]; };

static __device__ __forceinline__ __bf16 f2bf(float f) {
    union { float f; unsigned u; } in; in.f = f;
    unsigned r = (in.u + 0x7FFFu + ((in.u >> 16) & 1u)) >> 16;   // RNE
    union { unsigned short s; __bf16 b; } out; out.s = (unsigned short)r;
    return out.b;
}
static __device__ __forceinline__ float sigf(float x) { return 1.f / (1.f + expf(-x)); }

// ---- CDNA5 async global->LDS copy (ASYNCcnt path) -------------------------
// clang-22 signature (from diagnostic): (AS1 v4i* gsrc, AS3 v4i* ldst, Iint, Iint)
typedef __attribute__((address_space(1))) v4i g_v4i;
typedef __attribute__((address_space(3))) v4i l_v4i;

static __device__ __forceinline__ void async_copy_b128(const __bf16* g, __bf16* l) {
#if __has_builtin(__builtin_amdgcn_global_load_async_to_lds_b128)
    __builtin_amdgcn_global_load_async_to_lds_b128((g_v4i*)g, (l_v4i*)l, 0, 0);
#else
    asm volatile("global_load_async_to_lds_b128 %0, %1, off"
                 :: "v"((unsigned)(size_t)l), "v"(g) : "memory");
#endif
}
static __device__ __forceinline__ void wait_async0() {
#if __has_builtin(__builtin_amdgcn_s_wait_asynccnt)
    __builtin_amdgcn_s_wait_asynccnt(0);
#else
    asm volatile("s_wait_asynccnt 0x0" ::: "memory");
#endif
}

#define LDT 40   // LDS row stride (bf16 elems): 80B rows to stagger banks

// ---------------------------------------------------------------------------
// Async-staged WMMA GEMM: C[M,N] = A[M,K](bf16) * B[N,K](bf16)^T + bias[N]
// 256 threads / 8 waves, tile 128x128, K step 32, double-buffered LDS tiles.
// Requires: M % 128 == 0, K % 32 == 0; handles ragged N via dump slot.
// ---------------------------------------------------------------------------
__global__ __launch_bounds__(256) void gemm_wmma_async(
    const __bf16* __restrict__ A, int lda,
    const __bf16* __restrict__ B, int ldb,
    const float* __restrict__ bias,
    float* __restrict__ C, int ldc,
    int M, int N, int K)
{
    __shared__ alignas(16) __bf16 As[2][128 * LDT];
    __shared__ alignas(16) __bf16 Bs[2][128 * LDT];
    __shared__ alignas(16) __bf16 dump[8];

    const int tid  = threadIdx.x;
    const int wave = tid >> 5, lane = tid & 31;
    const int hf   = lane >> 4, l16 = lane & 15;
    const int m0 = blockIdx.y * 128, n0 = blockIdx.x * 128;
    const int rowOff = (wave >> 1) * 32;
    const int colOff = (wave & 1) * 64;

    // staging assignment: thread -> (row 0..127, 16B chunk 0..1) of each tile
    const int srow = tid >> 1;
    const int sc   = (tid & 1) * 8;            // bf16 element offset (0 / 8)
    const bool bok = (n0 + srow) < N;
    const __bf16* gA = A + (size_t)(m0 + srow) * lda + sc;
    const __bf16* gB = bok ? (B + (size_t)(n0 + srow) * ldb + sc) : B;

    v8f acc[2][4];
    for (int mi = 0; mi < 2; ++mi)
        for (int ni = 0; ni < 4; ++ni)
            for (int r = 0; r < 8; ++r) acc[mi][ni][r] = 0.f;

    const int nk = K >> 5;
    // prologue: stage K-chunk 0 into buffer 0
    async_copy_b128(gA, &As[0][srow * LDT + sc]);
    async_copy_b128(gB, bok ? &Bs[0][srow * LDT + sc] : dump);

    for (int i = 0; i < nk; ++i) {
        wait_async0();
        __syncthreads();                       // tiles[cur] visible to all waves
        const int cur = i & 1;
        if (i + 1 < nk) {                      // overlap next stage with WMMA
            const int k0 = (i + 1) << 5;
            async_copy_b128(gA + k0, &As[1 - cur][srow * LDT + sc]);
            async_copy_b128(bok ? gB + k0 : gB, bok ? &Bs[1 - cur][srow * LDT + sc] : dump);
        }
        FragBF fa[2], fb[4];
        for (int mi = 0; mi < 2; ++mi) {
            // A 16x32: lane = row; VGPR0-3 K=8*half..+7, VGPR4-7 K=16+8*half..+7
            const __bf16* p = &As[cur][(rowOff + mi * 16 + l16) * LDT + hf * 8];
            fa[mi].q[0] = *(const U4*)p;
            fa[mi].q[1] = *(const U4*)(p + 16);
        }
        for (int ni = 0; ni < 4; ++ni) {
            // B 32x16: lane = col; VGPR0-3 K=8*half..+7, VGPR4-7 K=16+8*half..+7
            const __bf16* p = &Bs[cur][(colOff + ni * 16 + l16) * LDT + hf * 8];
            fb[ni].q[0] = *(const U4*)p;
            fb[ni].q[1] = *(const U4*)(p + 16);
        }
        for (int mi = 0; mi < 2; ++mi)
            for (int ni = 0; ni < 4; ++ni)
                acc[mi][ni] = __builtin_amdgcn_wmma_f32_16x16x32_bf16(
                    false, fa[mi].v, false, fb[ni].v,
                    (short)0, acc[mi][ni], false, false);
    }

    for (int mi = 0; mi < 2; ++mi)
        for (int ni = 0; ni < 4; ++ni) {
            int col = n0 + colOff + ni * 16 + l16;
            if (col >= N) continue;
            float bv = bias ? bias[col] : 0.f;
            for (int r = 0; r < 8; ++r) {
                int row = m0 + rowOff + mi * 16 + hf * 8 + r;
                if (row < M)
                    C[(size_t)row * ldc + col] = acc[mi][ni][r] + bv;
            }
        }
}

// ---------------------------------------------------------------------------
// Convert-staging WMMA GEMM (f32 A/B -> bf16 in LDS); used once for
// f_att = feats @ Wp^T. Optionally mirrors C into bf16 (for att_f's A input).
// ---------------------------------------------------------------------------
__global__ __launch_bounds__(256) void gemm_wmma_cvt(
    const float* __restrict__ A, int lda,
    const float* __restrict__ B, int ldb,
    const float* __restrict__ bias,
    float* __restrict__ C, __bf16* __restrict__ Cbf, int ldc,
    int M, int N, int K)
{
    __shared__ alignas(16) __bf16 As[128 * LDT];
    __shared__ alignas(16) __bf16 Bs[128 * LDT];

    const int tid  = threadIdx.x;
    const int wave = tid >> 5, lane = tid & 31;
    const int hf   = lane >> 4, l16 = lane & 15;
    const int m0 = blockIdx.y * 128, n0 = blockIdx.x * 128;
    const int rowOff = (wave >> 1) * 32;
    const int colOff = (wave & 1) * 64;

    v8f acc[2][4];
    for (int mi = 0; mi < 2; ++mi)
        for (int ni = 0; ni < 4; ++ni)
            for (int r = 0; r < 8; ++r) acc[mi][ni][r] = 0.f;

    const int lr = tid >> 3;
    const int kb = (tid & 7) * 4;

    for (int k0 = 0; k0 < K; k0 += 32) {
        for (int it = 0; it < 4; ++it) {
            int row = it * 32 + lr;
            float4 v = make_float4(0.f, 0.f, 0.f, 0.f);
            if (m0 + row < M)
                v = *(const float4*)(A + (size_t)(m0 + row) * lda + k0 + kb);
            __bf16* d = &As[row * LDT + kb];
            d[0] = f2bf(v.x); d[1] = f2bf(v.y); d[2] = f2bf(v.z); d[3] = f2bf(v.w);
        }
        for (int it = 0; it < 4; ++it) {
            int row = it * 32 + lr;
            float4 v = make_float4(0.f, 0.f, 0.f, 0.f);
            if (n0 + row < N)
                v = *(const float4*)(B + (size_t)(n0 + row) * ldb + k0 + kb);
            __bf16* d = &Bs[row * LDT + kb];
            d[0] = f2bf(v.x); d[1] = f2bf(v.y); d[2] = f2bf(v.z); d[3] = f2bf(v.w);
        }
        __syncthreads();
        FragBF fa[2], fb[4];
        for (int mi = 0; mi < 2; ++mi) {
            const __bf16* p = &As[(rowOff + mi * 16 + l16) * LDT + hf * 8];
            fa[mi].q[0] = *(const U4*)p;
            fa[mi].q[1] = *(const U4*)(p + 16);
        }
        for (int ni = 0; ni < 4; ++ni) {
            const __bf16* p = &Bs[(colOff + ni * 16 + l16) * LDT + hf * 8];
            fb[ni].q[0] = *(const U4*)p;
            fb[ni].q[1] = *(const U4*)(p + 16);
        }
        for (int mi = 0; mi < 2; ++mi)
            for (int ni = 0; ni < 4; ++ni)
                acc[mi][ni] = __builtin_amdgcn_wmma_f32_16x16x32_bf16(
                    false, fa[mi].v, false, fb[ni].v,
                    (short)0, acc[mi][ni], false, false);
        __syncthreads();
    }

    for (int mi = 0; mi < 2; ++mi)
        for (int ni = 0; ni < 4; ++ni) {
            int col = n0 + colOff + ni * 16 + l16;
            if (col >= N) continue;
            float bv = bias ? bias[col] : 0.f;
            for (int r = 0; r < 8; ++r) {
                int row = m0 + rowOff + mi * 16 + hf * 8 + r;
                if (row < M) {
                    float v = acc[mi][ni][r] + bv;
                    C[(size_t)row * ldc + col] = v;
                    if (Cbf) Cbf[(size_t)row * ldc + col] = f2bf(v);
                }
            }
        }
}

// ---------------------------------------------------------------------------
// Elementwise / small kernels (activations written directly as bf16)
// ---------------------------------------------------------------------------
__global__ void lstm_cell_kernel(const float* __restrict__ g, float* __restrict__ c,
                                 __bf16* __restrict__ dst1, int ld1,
                                 __bf16* __restrict__ dst2, int ld2)
{
    int idx = blockIdx.x * blockDim.x + threadIdx.x;
    if (idx >= 512 * 512) return;
    int b = idx >> 9, j = idx & 511;
    const float* gr = g + (size_t)b * 2048;
    float i  = sigf(gr[j]);
    float f  = sigf(gr[512 + j]);
    float gg = tanhf(gr[1024 + j]);
    float o  = sigf(gr[1536 + j]);
    float cn = f * c[idx] + i * gg;
    c[idx] = cn;
    __bf16 h = f2bf(o * tanhf(cn));
    dst1[(size_t)b * ld1 + j] = h;
    if (dst2) dst2[(size_t)b * ld2 + j] = h;
}

__global__ __launch_bounds__(128) void att_e_kernel(
    const float* __restrict__ att_f, const float* __restrict__ ha,
    const float* __restrict__ wa2, const float* __restrict__ ba2,
    float* __restrict__ e)
{
    int bk = blockIdx.x;          // b*36 + k
    int b  = bk / 36;
    int t  = threadIdx.x;
    float s = 0.f;
    for (int j = 0; j < 4; ++j) {
        int a = t + j * 128;
        s += tanhf(att_f[(size_t)bk * 512 + a] + ha[(size_t)b * 512 + a]) * wa2[a];
    }
    __shared__ float red[128];
    red[t] = s; __syncthreads();
    for (int off = 64; off > 0; off >>= 1) {
        if (t < off) red[t] += red[t + off];
        __syncthreads();
    }
    if (t == 0) e[bk] = red[0] + ba2[0];
}

__global__ __launch_bounds__(512) void softmax_ctx_kernel(
    const float* __restrict__ e, const float* __restrict__ f_att,
    __bf16* __restrict__ ctx_dst, int ldd)
{
    int b = blockIdx.x, t = threadIdx.x;
    __shared__ float es[36];
    if (t < 36) es[t] = e[b * 36 + t];
    __syncthreads();
    float mx = es[0];
    for (int k = 1; k < 36; ++k) mx = fmaxf(mx, es[k]);
    float den = 0.f;
    for (int k = 0; k < 36; ++k) den += expf(es[k] - mx);
    float s = 0.f;
    for (int k = 0; k < 36; ++k)
        s += expf(es[k] - mx) * f_att[((size_t)b * 36 + k) * 512 + t];
    ctx_dst[(size_t)b * ldd + t] = f2bf(s / den);
}

__global__ __launch_bounds__(256) void argmax_embed_kernel(
    const float* __restrict__ logits, const float* __restrict__ E,
    int* __restrict__ ids, int step, __bf16* __restrict__ x_dst, int ld1)
{
    int b = blockIdx.x, t = threadIdx.x;
    const float* lr = logits + (size_t)b * 10000;
    float bv = -3.4e38f; int bi = 0x7fffffff;
    for (int j = t; j < 10000; j += 256) {
        float v = lr[j];
        if (v > bv) { bv = v; bi = j; }
    }
    __shared__ float rv[256];
    __shared__ int   ri[256];
    rv[t] = bv; ri[t] = bi; __syncthreads();
    for (int off = 128; off > 0; off >>= 1) {
        if (t < off) {
            if (rv[t + off] > rv[t] || (rv[t + off] == rv[t] && ri[t + off] < ri[t])) {
                rv[t] = rv[t + off]; ri[t] = ri[t + off];
            }
        }
        __syncthreads();
    }
    int idx = ri[0];
    if (t == 0) ids[b * 19 + step] = idx;              // ids.T -> (B,19)
    for (int j = t; j < 300; j += 256)                 // x = E[argmax]
        x_dst[(size_t)b * ld1 + j] = f2bf(E[(size_t)idx * 300 + j]);
}

__global__ __launch_bounds__(512) void f_mean_kernel(
    const float* __restrict__ f_att, __bf16* __restrict__ dst, int ld)
{
    int b = blockIdx.x, a = threadIdx.x;
    float s = 0.f;
    for (int k = 0; k < 36; ++k) s += f_att[((size_t)b * 36 + k) * 512 + a];
    dst[(size_t)b * ld + a] = f2bf(s * (1.f / 36.f));
}

// pack [Wih1 | Whh1] -> bf16 (2048 x 1344), in1 layout [x 0:304 | f_mean 304:816 | h1 816:1328 | pad]
__global__ void pack_w1_kernel(const float* __restrict__ Wih1,
                               const float* __restrict__ Whh1,
                               __bf16* __restrict__ W1cat)
{
    int r = blockIdx.x;
    for (int c = threadIdx.x; c < 1344; c += blockDim.x) {
        float v = 0.f;
        if (c < 300)        v = Wih1[(size_t)r * 812 + c];          // x cols
        else if (c < 304)   v = 0.f;                                // x pad
        else if (c < 816)   v = Wih1[(size_t)r * 812 + (c - 4)];    // f_mean cols
        else if (c < 1328)  v = Whh1[(size_t)r * 512 + (c - 816)];  // h1 cols
        W1cat[(size_t)r * 1344 + c] = f2bf(v);
    }
}
// pack [Wih2 | Whh2] -> bf16 (2048 x 1536)
__global__ void pack_w2_kernel(const float* __restrict__ Wih2,
                               const float* __restrict__ Whh2,
                               __bf16* __restrict__ W2cat)
{
    int r = blockIdx.x;
    for (int c = threadIdx.x; c < 1536; c += blockDim.x) {
        float v = (c < 1024) ? Wih2[(size_t)r * 1024 + c]
                             : Whh2[(size_t)r * 512 + (c - 1024)];
        W2cat[(size_t)r * 1536 + c] = f2bf(v);
    }
}

__global__ void wa1_split_kernel(const float* __restrict__ Wa1,
                                 __bf16* __restrict__ Wh, __bf16* __restrict__ Wf)
{
    int idx = blockIdx.x * blockDim.x + threadIdx.x;
    if (idx >= 512 * 512) return;
    int r = idx >> 9, c = idx & 511;
    Wh[idx] = f2bf(Wa1[(size_t)r * 1024 + c]);
    Wf[idx] = f2bf(Wa1[(size_t)r * 1024 + 512 + c]);
}

__global__ void cvt_bf16_kernel(const float* __restrict__ src,
                                __bf16* __restrict__ dst, int n)
{
    int idx = blockIdx.x * blockDim.x + threadIdx.x;
    if (idx < n) dst[idx] = f2bf(src[idx]);
}

__global__ void bias_init_kernel(const float* bih1, const float* bhh1,
                                 const float* bih2, const float* bhh2,
                                 float* b1, float* b2)
{
    int j = blockIdx.x * blockDim.x + threadIdx.x;
    if (j < 2048) { b1[j] = bih1[j] + bhh1[j]; b2[j] = bih2[j] + bhh2[j]; }
}

// zero states; x0 = bf16(E[BOS=1]) into in1 cols [0,300)
__global__ void state_init_kernel(const float* __restrict__ E,
                                  __bf16* in1, __bf16* in2, float* c1, float* c2)
{
    int idx = blockIdx.x * blockDim.x + threadIdx.x;
    if (idx < 512 * 1344) {
        int c = idx % 1344;
        in1[idx] = (c < 300) ? f2bf(E[300 + c]) : f2bf(0.f);
    }
    if (idx < 512 * 1536) in2[idx] = f2bf(0.f);
    if (idx < 512 * 512) { c1[idx] = 0.f; c2[idx] = 0.f; }
}

// ---------------------------------------------------------------------------
// Host orchestration
// ---------------------------------------------------------------------------
extern "C" void kernel_launch(void* const* d_in, const int* in_sizes, int n_in,
                              void* d_out, int out_size, void* d_ws, size_t ws_size,
                              hipStream_t stream)
{
    const float* feats = (const float*)d_in[0];
    const float* Wp    = (const float*)d_in[1];
    const float* bp    = (const float*)d_in[2];
    const float* E     = (const float*)d_in[3];
    const float* Wih1  = (const float*)d_in[4];
    const float* Whh1  = (const float*)d_in[5];
    const float* bih1  = (const float*)d_in[6];
    const float* bhh1  = (const float*)d_in[7];
    const float* Wih2  = (const float*)d_in[8];
    const float* Whh2  = (const float*)d_in[9];
    const float* bih2  = (const float*)d_in[10];
    const float* bhh2  = (const float*)d_in[11];
    const float* Wa1   = (const float*)d_in[12];
    const float* ba1   = (const float*)d_in[13];
    const float* wa2   = (const float*)d_in[14];
    const float* ba2   = (const float*)d_in[15];
    const float* Wo    = (const float*)d_in[16];
    const float* bo    = (const float*)d_in[17];

    float* ws = (float*)d_ws;
    // f32 workspace (float offsets)
    float* f_att  = ws + 0;            // 18432*512
    float* att_f  = ws + 9437184;      // 18432*512
    float* c1     = ws + 18874368;     // 512*512
    float* c2     = ws + 19136512;     // 512*512
    float* gbuf   = ws + 19398656;     // 512*2048
    float* ha     = ws + 20447232;     // 512*512
    float* ebuf   = ws + 20709376;     // 512*36
    float* b1sum  = ws + 20727808;     // 2048
    float* b2sum  = ws + 20729856;     // 2048
    float* logits = ws + 20731904;     // 512*10000
    // bf16 workspace (carved at float offsets, all 16B aligned)
    __bf16* W1cat_bf = (__bf16*)(ws + 25851904);   // 2048*1344
    __bf16* W2cat_bf = (__bf16*)(ws + 27228160);   // 2048*1536
    __bf16* Wa1h_bf  = (__bf16*)(ws + 28801024);   // 512*512
    __bf16* Wa1f_bf  = (__bf16*)(ws + 28932096);   // 512*512
    __bf16* Wo_bf    = (__bf16*)(ws + 29063168);   // 10000*512
    __bf16* fatt_bf  = (__bf16*)(ws + 31623168);   // 18432*512
    __bf16* in1_bf   = (__bf16*)(ws + 36341760);   // 512*1344
    __bf16* in2_bf   = (__bf16*)(ws + 36685824);   // 512*1536  (end: 37079040 f)

    int* ids = (int*)d_out;            // (B, 19) argmax token ids

    // ---- one-time packing / init ----
    pack_w1_kernel<<<2048, 256, 0, stream>>>(Wih1, Whh1, W1cat_bf);
    pack_w2_kernel<<<2048, 256, 0, stream>>>(Wih2, Whh2, W2cat_bf);
    wa1_split_kernel<<<1024, 256, 0, stream>>>(Wa1, Wa1h_bf, Wa1f_bf);
    cvt_bf16_kernel<<<(5120000 + 255) / 256, 256, 0, stream>>>(Wo, Wo_bf, 5120000);
    bias_init_kernel<<<8, 256, 0, stream>>>(bih1, bhh1, bih2, bhh2, b1sum, b2sum);
    state_init_kernel<<<(512 * 1536 + 255) / 256, 256, 0, stream>>>(E, in1_bf, in2_bf, c1, c2);

    // ---- precompute: f_att = feats @ Wp^T + bp (f32+bf16 mirror) ----
    gemm_wmma_cvt<<<dim3(4, 144), 256, 0, stream>>>(
        feats, 2048, Wp, 2048, bp, f_att, fatt_bf, 512, 18432, 512, 2048);
    f_mean_kernel<<<512, 512, 0, stream>>>(f_att, in1_bf + 304, 1344);
    // att_f = f_att @ Wa1_f^T + ba1
    gemm_wmma_async<<<dim3(4, 144), 256, 0, stream>>>(
        fatt_bf, 512, Wa1f_bf, 512, ba1, att_f, 512, 18432, 512, 512);

    // ---- 19 greedy decode steps ----
    for (int step = 0; step < 19; ++step) {
        // g1 = [x|f_mean|h1] @ [Wih1|Whh1]^T + (bih1+bhh1)
        gemm_wmma_async<<<dim3(16, 4), 256, 0, stream>>>(
            in1_bf, 1344, W1cat_bf, 1344, b1sum, gbuf, 2048, 512, 2048, 1344);
        lstm_cell_kernel<<<1024, 256, 0, stream>>>(
            gbuf, c1, in1_bf + 816, 1344, in2_bf, 1536);    // h1 -> in1, in2
        // ha = h1 @ Wa1_h^T
        gemm_wmma_async<<<dim3(4, 4), 256, 0, stream>>>(
            in1_bf + 816, 1344, Wa1h_bf, 512, nullptr, ha, 512, 512, 512, 512);
        att_e_kernel<<<18432, 128, 0, stream>>>(att_f, ha, wa2, ba2, ebuf);
        softmax_ctx_kernel<<<512, 512, 0, stream>>>(ebuf, f_att, in2_bf + 512, 1536);
        // g2 = [h1|ctx|h2] @ [Wih2|Whh2]^T + (bih2+bhh2)
        gemm_wmma_async<<<dim3(16, 4), 256, 0, stream>>>(
            in2_bf, 1536, W2cat_bf, 1536, b2sum, gbuf, 2048, 512, 2048, 1536);
        lstm_cell_kernel<<<1024, 256, 0, stream>>>(
            gbuf, c2, in2_bf + 1024, 1536, nullptr, 0);     // h2 -> in2
        // logits = h2 @ Wo^T + bo
        gemm_wmma_async<<<dim3(79, 4), 256, 0, stream>>>(
            in2_bf + 1024, 1536, Wo_bf, 512, bo, logits, 10000, 512, 10000, 512);
        argmax_embed_kernel<<<512, 256, 0, stream>>>(
            logits, E, ids, step, in1_bf, 1344);
    }
}
